// RNN_46445776339176
// MI455X (gfx1250) — matrix operand order repaired and checked by reading, not compiled, not run on hardware
//
#include <hip/hip_runtime.h>

typedef __attribute__((ext_vector_type(16))) _Float16 v16h;
typedef __attribute__((ext_vector_type(8)))  _Float16 v8h;
typedef __attribute__((ext_vector_type(8)))  float    v8f;
typedef __attribute__((ext_vector_type(4)))  float    v4f;

#define SEQ_LEN 2048
#define BATCH   64
#define HSZ     256

#if __has_builtin(__builtin_amdgcn_tanhf)
  #define TANH_F32(x) __builtin_amdgcn_tanhf(x)
#elif __has_builtin(__builtin_amdgcn_tanh_f32)
  #define TANH_F32(x) __builtin_amdgcn_tanh_f32(x)
#else
  #define TANH_F32(x) tanhf(x)
#endif

// =====================================================================
// Phase 1:  x_proj[r, n] = sum_k input[r, k] * w_ih[n, k] + (b_ih[n]+b_hh[n])
//   R = S*B = 131072 rows.  Block = 512 threads (16 waves), tile M=128 x N=256.
//   Wave w owns n-slice [16w, 16w+16), loops over 8 m-tiles of 16.
//   Memory-bound: 128 MB in + 128 MB out @ 23.3 TB/s.
// =====================================================================
__global__ __launch_bounds__(512)
void rnn_xproj_kernel(const float* __restrict__ x,     // [R, 256]
                      const float* __restrict__ w_ih,  // [256, 256] row-major [n, k]
                      const float* __restrict__ b_ih,
                      const float* __restrict__ b_hh,
                      float* __restrict__ xp)          // [R, 256]  (head of d_out)
{
    __shared__ __align__(16) _Float16 As[128 * 256];   // 64 KB f16 A tile

    const int  tid  = threadIdx.x;
    const int  wave = tid >> 5;
    const int  lane = tid & 31;
    const int  l16  = lane & 15;
    const int  lhi  = lane >> 4;                       // 0 or 1
    const long r0   = (long)blockIdx.x * 128;

    // cooperative, fully-coalesced load of the 128x256 f32 A tile -> f16 LDS
    {
        const float* g = x + r0 * 256;
        #pragma unroll
        for (int it = 0; it < 16; ++it) {
            int e = (it * 512 + tid) * 4;
            v4f v = *(const v4f*)(g + e);
            As[e + 0] = (_Float16)v.x;
            As[e + 1] = (_Float16)v.y;
            As[e + 2] = (_Float16)v.z;
            As[e + 3] = (_Float16)v.w;
        }
    }
    __syncthreads();

    const int n0 = wave * 16;
    v8f acc[8] = {};                                   // 8 m-tiles of 16x16 f32

    #pragma unroll
    for (int t = 0; t < 8; ++t) {                      // K chunks of 32
        // B fragment (32x16): lane%16 = n, element j = k (table order, matches A)
        v16h b;
        #pragma unroll
        for (int j = 0; j < 16; ++j)
            b[j] = (_Float16)w_ih[(n0 + l16) * 256 + t * 32 + lhi * 8 + j + (j >= 8 ? 8 : 0)];

        const int kb = t * 32 + lhi * 8;
        #pragma unroll
        for (int mt = 0; mt < 8; ++mt) {
            union { v16h v; v8h h[2]; } a;
            const _Float16* ap = &As[(mt * 16 + l16) * 256 + kb];
            a.h[0] = *(const v8h*)(ap);                // K = kb .. kb+7
            a.h[1] = *(const v8h*)(ap + 16);           // K = kb+16 .. kb+23
            acc[mt] = __builtin_amdgcn_wmma_f32_16x16x32_f16(
                false, a.v, false, b, (short)0, acc[mt], false, false);
        }
    }

    const float bias = b_ih[n0 + l16] + b_hh[n0 + l16];
    #pragma unroll
    for (int mt = 0; mt < 8; ++mt) {
        #pragma unroll
        for (int v = 0; v < 8; ++v) {
            long row = r0 + mt * 16 + lhi * 8 + v;     // C layout: m = v + 8*lhi
            xp[row * 256 + n0 + l16] = acc[mt][v] + bias;
        }
    }
}

// =====================================================================
// Phase 2:  for s in 0..S-1:  h = tanh(xp[s] + h @ w_hh^T)
//   Batch-independent recurrence -> 4 blocks of 16 batch rows.
//   Block = 512 threads (16 waves); wave w owns n0 = 16w and keeps its
//   w_hh B-fragments resident in 64 VGPRs.  h is double-buffered f16 LDS;
//   one barrier per step.  xp is software-pipelined one step ahead in
//   registers, plus global_prefetch 6 steps ahead into L2.  Two
//   interleaved WMMA accumulator chains cut the RAW dependency depth.
// =====================================================================
__global__ __launch_bounds__(512)
void rnn_recur_kernel(const float* __restrict__ w_hh,  // [256, 256] row-major [n, k]
                      float* __restrict__ out,         // [S*B*256] x_proj -> output
                      float* __restrict__ hfin)        // [B*256] final hidden state
{
    __shared__ __align__(16) _Float16 hbuf[2][16 * 256];

    const int tid  = threadIdx.x;
    const int wave = tid >> 5;
    const int lane = tid & 31;
    const int l16  = lane & 15;
    const int lhi  = lane >> 4;
    const int rb0  = blockIdx.x * 16;                  // batch-row base
    const int n0   = wave * 16;

    // persistent B fragments of w_hh (whole K=256 for this wave's n-slice)
    v16h bh[8];
    #pragma unroll
    for (int t = 0; t < 8; ++t) {
        #pragma unroll
        for (int j = 0; j < 16; ++j)
            bh[t][j] = (_Float16)w_hh[(n0 + l16) * 256 + t * 32 + lhi * 8 + j + (j >= 8 ? 8 : 0)];
    }

    // h0 = 0
    for (int e = tid; e < 16 * 256; e += 512) hbuf[0][e] = (_Float16)0.0f;
    __syncthreads();

    // per-lane element offset inside a step tile (C layout: m = v + 8*lhi)
    const int coff = (lhi * 8) * 256 + n0 + l16;       // + v*256 per element
    const long tile0 = (long)rb0 * 256;                // within step s: + s*B*256

    // pipeline: xp[0] into registers
    v8f xpc;
    {
        const float* p = out + tile0 + coff;
        #pragma unroll
        for (int v = 0; v < 8; ++v) xpc[v] = p[v * 256];
    }

    for (int s = 0; s < SEQ_LEN; ++s) {
        const int cur = s & 1, nxt = cur ^ 1;
        float* xps = out + (long)s * (BATCH * 256) + tile0;

        // deep prefetch: step s+6 tile (16 rows x 1 KB) into L2
        if (tid < 128) {
            const int sp = (s + 6 < SEQ_LEN) ? s + 6 : s;
            __builtin_prefetch(out + (long)sp * (BATCH * 256) + tile0 + tid * 32, 0, 1);
        }

        // register prefetch of next step's xp (overlaps the WMMA chain)
        v8f xpn;
        {
            const int sn = (s + 1 < SEQ_LEN) ? s + 1 : s;
            const float* p = out + (long)sn * (BATCH * 256) + tile0 + coff;
            #pragma unroll
            for (int v = 0; v < 8; ++v) xpn[v] = p[v * 256];
        }

        // two interleaved accumulator chains: acc0 = xp + sum(even k), acc1 = sum(odd k)
        v8f acc0 = xpc;
        v8f acc1 = {};
        #pragma unroll
        for (int t = 0; t < 8; ++t) {
            const int kb = t * 32 + lhi * 8;
            union { v16h v; v8h h[2]; } a;
            const _Float16* ap = &hbuf[cur][l16 * 256 + kb];
            a.h[0] = *(const v8h*)(ap);
            a.h[1] = *(const v8h*)(ap + 16);
            if (t & 1)
                acc1 = __builtin_amdgcn_wmma_f32_16x16x32_f16(
                    false, a.v, false, bh[t], (short)0, acc1, false, false);
            else
                acc0 = __builtin_amdgcn_wmma_f32_16x16x32_f16(
                    false, a.v, false, bh[t], (short)0, acc0, false, false);
        }

        // tanh (hardware trans op), write output + next-step h
        #pragma unroll
        for (int v = 0; v < 8; ++v) {
            float hval = TANH_F32(acc0[v] + acc1[v]);
            xps[coff + v * 256]            = hval;
            hbuf[nxt][coff + v * 256]      = (_Float16)hval;
            if (s == SEQ_LEN - 1)
                hfin[tile0 + coff + v * 256] = hval;   // f32 final hidden state
        }
        __syncthreads();
        xpc = xpn;
    }
}

extern "C" void kernel_launch(void* const* d_in, const int* in_sizes, int n_in,
                              void* d_out, int out_size, void* d_ws, size_t ws_size,
                              hipStream_t stream) {
    (void)in_sizes; (void)n_in; (void)d_ws; (void)ws_size; (void)out_size;
    const float* x    = (const float*)d_in[0];
    const float* w_ih = (const float*)d_in[1];
    const float* b_ih = (const float*)d_in[2];
    const float* w_hh = (const float*)d_in[3];
    const float* b_hh = (const float*)d_in[4];
    float* out = (float*)d_out;

    const int R = SEQ_LEN * BATCH;                     // 131072 rows
    rnn_xproj_kernel<<<R / 128, 512, 0, stream>>>(x, w_ih, b_ih, b_hh, out);
    rnn_recur_kernel<<<BATCH / 16, 512, 0, stream>>>(
        w_hh, out, out + (long)R * HSZ);
}
